// Attention_38920993636911
// MI455X (gfx1250) — matrix operand order repaired and checked by reading, not compiled, not run on hardware
//
#include <hip/hip_runtime.h>
#include <hip/hip_bf16.h>
#include <cstdint>

// Problem constants (from reference)
#define MB 8
#define K_LEN 2048
#define Q_LEN 512
#define EMBED 1024
#define N_HEAD 16
#define HIDDEN 64

static_assert(K_LEN == 2048, "phase-3 bit tricks assume K_LEN=2048");

typedef __attribute__((ext_vector_type(16))) __bf16    v16bf;
typedef __attribute__((ext_vector_type(8)))  float     v8f;
typedef __attribute__((ext_vector_type(4)))  unsigned  v4u;
typedef __attribute__((ext_vector_type(4)))  float     v4f;
typedef __attribute__((ext_vector_type(4)))  int       v4i;
typedef __attribute__((ext_vector_type(8)))  int       v8i;

#if __has_builtin(__builtin_amdgcn_tensor_load_to_lds) && \
    __has_builtin(__builtin_amdgcn_s_wait_tensorcnt)
#define HAVE_TDM 1
#endif

#if defined(__HIP_DEVICE_COMPILE__) && !defined(HAVE_TDM)
#warning "gfx1250 TDM builtin NOT available on this toolchain; using cooperative LDS copy fallback"
#endif

__device__ __forceinline__ unsigned bf16_1(float f) {
  unsigned u = __float_as_uint(f);
  unsigned r = u + 0x7FFFu + ((u >> 16) & 1u);  // round-to-nearest-even
  return r >> 16;
}
__device__ __forceinline__ unsigned pack_bf16(float lo, float hi) {
  return bf16_1(lo) | (bf16_1(hi) << 16);
}

__device__ __forceinline__ v8f wmma_bf16(v16bf a, v16bf b, v8f c) {
  return __builtin_amdgcn_wmma_f32_16x16x32_bf16(
      false, a, false, b, (short)0, c, false, false);
}

// ---------------- fragment loaders ----------------
// A fragment (16x32), bf16 row-major source, K contiguous.
__device__ __forceinline__ v16bf load_a_frag(const unsigned short* base, int ld, int lane) {
  int m = lane & 15, kb = (lane >> 4) * 8;
  const unsigned short* r = base + (long)m * ld + kb;
  union U { v4u q[2]; v16bf v; } u;
  u.q[0] = *reinterpret_cast<const v4u*>(r);
  u.q[1] = *reinterpret_cast<const v4u*>(r + 16);
  return u.v;
}

// B fragment (32x16), bf16 source stored N-major with K contiguous.
__device__ __forceinline__ v16bf load_b_frag(const unsigned short* base, int ld, int lane) {
  int n = lane & 15, ko = (lane >> 4) * 8;
  const v4u* p = reinterpret_cast<const v4u*>(base + (long)n * ld + ko);
  union U { v4u q[2]; v16bf v; } u;
  u.q[0] = p[0];
  u.q[1] = p[2];
  return u.v;
}

// A fragment from f32 row-major source (convert to bf16 in-register).
__device__ __forceinline__ v16bf load_a_frag_f32g(const float* base, int ld, int lane) {
  int m = lane & 15, kb = (lane >> 4) * 8;
  const v4f* p  = reinterpret_cast<const v4f*>(base + (long)m * ld + kb);
  const v4f* p2 = reinterpret_cast<const v4f*>(base + (long)m * ld + kb + 16);
  v4f a0 = p[0], a1 = p[1], b0 = p2[0], b1 = p2[1];
  union U { unsigned u[8]; v16bf v; } r;
  r.u[0] = pack_bf16(a0[0], a0[1]); r.u[1] = pack_bf16(a0[2], a0[3]);
  r.u[2] = pack_bf16(a1[0], a1[1]); r.u[3] = pack_bf16(a1[2], a1[3]);
  r.u[4] = pack_bf16(b0[0], b0[1]); r.u[5] = pack_bf16(b0[2], b0[3]);
  r.u[6] = pack_bf16(b1[0], b1[1]); r.u[7] = pack_bf16(b1[2], b1[3]);
  return r.v;
}

// A fragment from an f32 LDS strip.
__device__ __forceinline__ v16bf load_a_frag_lds(const float* base, int ld, int lane, int k0) {
  int m = lane & 15, kb = (lane >> 4) * 8;
  const float* p = base + (long)m * ld + k0 + kb;
  union U { unsigned u[8]; v16bf v; } r;
#pragma unroll
  for (int i = 0; i < 4; ++i) r.u[i] = pack_bf16(p[2 * i], p[2 * i + 1]);
#pragma unroll
  for (int i = 0; i < 4; ++i) r.u[4 + i] = pack_bf16(p[16 + 2 * i], p[16 + 2 * i + 1]);
  return r.v;
}

// ---------------- padded LDS weight panel (64 rows x 1024 bf16) ----------------
// Row stride 2080 B (16 B pad per 1024 B) -> n-rows staggered across banks.
#define WLDS_ROW 1040  // elements
#define WLDS_ELEMS (HIDDEN * WLDS_ROW)

__device__ __forceinline__ int lds_pad_off(int n, int k) {
  return n * WLDS_ROW + k + ((k >> 9) << 3);
}

// B fragment from the padded LDS panel.
__device__ __forceinline__ v16bf load_b_frag_ldsw(const unsigned short* lds_w, int row0,
                                                  int k0, int lane) {
  int n = row0 + (lane & 15);
  int ko = k0 + ((lane >> 4) << 3);
  const v4u* p = reinterpret_cast<const v4u*>(lds_w + lds_pad_off(n, ko));
  union U { v4u q[2]; v16bf v; } u;
  u.q[0] = p[0];
  u.q[1] = p[2];
  return u.v;
}

// Stage a 64x1024 bf16 weight panel into LDS (TDM DMA with pad, or cooperative copy).
__device__ __forceinline__ void stage_weights_to_lds(unsigned short* lds_w,
                                                     const unsigned short* wb) {
#if defined(HAVE_TDM)
  if (threadIdx.x < 32) {  // one wave issues the tensor DMA
    unsigned long long ga = (unsigned long long)(uintptr_t)wb;
    unsigned la = (unsigned)(uintptr_t)lds_w;  // low 32 bits = LDS byte offset
    v4u g0;
    g0[0] = 1u;                                   // count=1 (valid user descriptor)
    g0[1] = la;                                   // lds_addr
    g0[2] = (unsigned)(ga & 0xFFFFFFFFu);         // global_addr[31:0]
    g0[3] = (unsigned)((ga >> 32) & 0x01FFFFFFu) | (2u << 30);  // addr[56:32] | type=2
    v8i g1;
    g1[0] = (int)((1u << 16) |                    // data_size = 2 bytes
                  (1u << 20) |                    // pad_enable
                  (7u << 22) |                    // pad_interval: 256 DW = 1024 B
                  (3u << 25));                    // pad_amount: 4 DW = 16 B
    g1[1] = (int)(1024u << 16);                   // tensor_dim0[15:0] = 1024
    g1[2] = (int)(64u << 16);                     // tensor_dim1[15:0] = 64
    g1[3] = (int)(1024u << 16);                   // tile_dim0 = 1024
    g1[4] = 64;                                   // tile_dim1 = 64, tile_dim2 = 0
    g1[5] = 1024;                                 // tensor_dim0_stride = 1024
    g1[6] = 0;
    g1[7] = 0;
    v4i z = {0, 0, 0, 0};
#if __clang_major__ >= 23
    v8i z8 = {0, 0, 0, 0, 0, 0, 0, 0};
    __builtin_amdgcn_tensor_load_to_lds(g0, g1, z, z, z8, 0);
#else
    __builtin_amdgcn_tensor_load_to_lds(g0, g1, z, z, 0);
#endif
    __builtin_amdgcn_s_wait_tensorcnt(0);
  }
#else
  for (int idx = threadIdx.x; idx < HIDDEN * (EMBED / 8); idx += blockDim.x) {
    int n = idx >> 7;           // 128 16-byte chunks per row
    int kc = (idx & 127) << 3;  // element offset in row
    *reinterpret_cast<v4u*>(lds_w + lds_pad_off(n, kc)) =
        *reinterpret_cast<const v4u*>(wb + (long)n * EMBED + kc);
  }
#endif
  __syncthreads();
}

// -------- Kernel 1: transpose + f32->bf16 convert.  w[h][R][C] -> wt[h][C][R]
__global__ void transpose_cvt_kernel(const float* __restrict__ w, unsigned short* __restrict__ wt,
                                     int H, int R, int C) {
  long total = (long)H * R * C;
  for (long i = blockIdx.x * (long)blockDim.x + threadIdx.x; i < total;
       i += (long)gridDim.x * blockDim.x) {
    long h   = i / ((long)R * C);
    long rem = i - h * (long)R * C;
    int  rr  = (int)(rem / C);
    int  cc  = (int)(rem % C);
    wt[h * (long)R * C + (long)cc * R + rr] = (unsigned short)bf16_1(w[i]);
  }
}

// -------- Kernel 2: X[h][b] = alpha * In[b] @ W[h]  (bf16 out; optional transposed copy)
template <bool WRITE_XT>
__global__ __launch_bounds__(256) void qk_proj_kernel(
    const float* __restrict__ In, const unsigned short* __restrict__ Wt,
    unsigned short* __restrict__ X, unsigned short* __restrict__ XT, int L, float alpha) {
  __shared__ unsigned short lds_w[WLDS_ELEMS];  // 130 KB weight panel
  const int m0 = blockIdx.x * 128;
  const int b = blockIdx.y, h = blockIdx.z;
  const int wave = threadIdx.x >> 5, lane = threadIdx.x & 31;

  stage_weights_to_lds(lds_w, Wt + (long)h * HIDDEN * EMBED);

  const float* arow = In + ((long)b * L + m0 + wave * 16) * EMBED;
  v8f acc[4] = {};
  v16bf a = load_a_frag_f32g(arow, EMBED, lane);
  for (int k0 = 0; k0 < EMBED; k0 += 32) {
    v16bf an = a;
    if (k0 + 32 < EMBED) an = load_a_frag_f32g(arow + k0 + 32, EMBED, lane);  // pipeline
    v16bf b0 = load_b_frag_ldsw(lds_w, 0, k0, lane);
    v16bf b1 = load_b_frag_ldsw(lds_w, 16, k0, lane);
    v16bf b2 = load_b_frag_ldsw(lds_w, 32, k0, lane);
    v16bf b3 = load_b_frag_ldsw(lds_w, 48, k0, lane);
    acc[0] = wmma_bf16(a, b0, acc[0]);
    acc[1] = wmma_bf16(a, b1, acc[1]);
    acc[2] = wmma_bf16(a, b2, acc[2]);
    acc[3] = wmma_bf16(a, b3, acc[3]);
    a = an;
  }
  const long hb = (long)h * MB + b;
  unsigned short* xr = X + (hb * L + m0 + wave * 16) * HIDDEN;
  const int n = lane & 15;
#pragma unroll
  for (int nt = 0; nt < 4; ++nt)
#pragma unroll
    for (int v = 0; v < 8; ++v) {
      int m = (lane >> 4) * 8 + v;
      unsigned short val = (unsigned short)bf16_1(alpha * acc[nt][v]);
      xr[(long)m * HIDDEN + nt * 16 + n] = val;
      if (WRITE_XT) XT[(hb * HIDDEN + nt * 16 + n) * (long)L + m0 + wave * 16 + m] = val;
    }
}

// -------- Kernel 3: attention. One block = (h, b, 16 q rows). 8 waves.
#define SROW (K_LEN + 8)  // padded score-row stride (floats) to stagger LDS banks

__global__ __launch_bounds__(256) void attn_kernel(
    const unsigned short* __restrict__ qx, const unsigned short* __restrict__ kx,
    const unsigned short* __restrict__ kxT, const int* __restrict__ memory_len,
    unsigned short* __restrict__ obuf, float* __restrict__ attn_out) {
  __shared__ float Sbuf[16][SROW];  // ~128.5 KB of the 320 KB WGP LDS
  __shared__ float red[16][16];
  __shared__ float rowmax[16];
  __shared__ float rowinv[16];
  __shared__ float Opart[2][4][16][16];

  const int q0 = blockIdx.x * 16;
  const int b = blockIdx.y, h = blockIdx.z;
  const int tid = threadIdx.x;
  const int wave = tid >> 5, lane = tid & 31;
  const long hb = (long)h * MB + b;
  const unsigned short* qrow = qx + (hb * Q_LEN + q0) * HIDDEN;
  const unsigned short* kbase = kx + hb * (long)K_LEN * HIDDEN;
  const int mlen = memory_len[b];

  // ---- Phase 1: S = qx @ kx^T (1/sqrt(d) pre-folded into qx), mask, stage in LDS
  v16bf a0 = load_a_frag(qrow + 0, HIDDEN, lane);
  v16bf a1 = load_a_frag(qrow + 32, HIDDEN, lane);
  // double-buffered B fragments: issue kt+8's loads before consuming kt's
  v16bf b0 = load_b_frag(kbase + (long)wave * 16 * HIDDEN, HIDDEN, lane);
  v16bf b1 = load_b_frag(kbase + (long)wave * 16 * HIDDEN + 32, HIDDEN, lane);
  for (int kt = wave; kt < K_LEN / 16; kt += 8) {
    v16bf nb0 = b0, nb1 = b1;
    if (kt + 8 < K_LEN / 16) {
      const unsigned short* nk = kbase + (long)(kt + 8) * 16 * HIDDEN;
      __builtin_prefetch(kbase + (long)(kt + 16) * 16 * HIDDEN + lane * 32, 0, 1);
      nb0 = load_b_frag(nk, HIDDEN, lane);
      nb1 = load_b_frag(nk + 32, HIDDEN, lane);
    }
    v8f acc = {};
    acc = wmma_bf16(a0, b0, acc);
    acc = wmma_bf16(a1, b1, acc);
    const int n = lane & 15;
    const int kpos = kt * 16 + n;
    const bool valid = kpos < mlen;
#pragma unroll
    for (int v = 0; v < 8; ++v) {
      int m = (lane >> 4) * 8 + v;
      Sbuf[m][kt * 16 + n] = valid ? acc[v] : -1.0e4f;
    }
    b0 = nb0;
    b1 = nb1;
  }
  __syncthreads();

  // ---- Phase 2: row max, then exp in-place + row sum (16 threads per row)
  const int r = tid & 15;
  const int ch = tid >> 4;
  float mx = -3.0e38f;
  for (int kk = ch * 128; kk < ch * 128 + 128; ++kk) mx = fmaxf(mx, Sbuf[r][kk]);
  red[r][ch] = mx;
  __syncthreads();
  if (tid < 16) {
    float m2 = -3.0e38f;
    for (int i = 0; i < 16; ++i) m2 = fmaxf(m2, red[tid][i]);
    rowmax[tid] = m2;
  }
  __syncthreads();
  const float rm = rowmax[r];
  float sum = 0.f;
  for (int kk = ch * 128; kk < ch * 128 + 128; ++kk) {
    float e = __expf(Sbuf[r][kk] - rm);
    Sbuf[r][kk] = e;
    sum += e;
  }
  red[r][ch] = sum;
  __syncthreads();
  if (tid < 16) {
    float s = 0.f;
    for (int i = 0; i < 16; ++i) s += red[tid][i];
    rowinv[tid] = 1.0f / s;
  }
  __syncthreads();

  // ---- Phase 3: coalesced attn write. Non-temporal: 512 MB streamed once, never
  // re-read by this op -> keep kx/kxT resident in the 192 MB L2 instead.
  float* arow_out = attn_out + (hb * Q_LEN + q0) * (long)K_LEN;
  for (int idx = tid; idx < 16 * K_LEN; idx += 256) {
    int rr = idx >> 11;
    int kk = idx & (K_LEN - 1);
    __builtin_nontemporal_store(Sbuf[rr][kk] * rowinv[rr], &arow_out[(long)rr * K_LEN + kk]);
  }

  // ---- Phase 4: O = attn @ kx via WMMA; wave -> (n-tile, K half), reduce halves in LDS
  const int ntile = wave & 3, half = wave >> 2;
  const unsigned short* ktb = kxT + (hb * HIDDEN + ntile * 16) * (long)K_LEN;
  v8f acc = {};
  v16bf bb = load_b_frag(ktb + half * 1024, K_LEN, lane);
  for (int kp0 = half * 1024; kp0 < half * 1024 + 1024; kp0 += 32) {
    v16bf nbb = bb;
    if (kp0 + 32 < half * 1024 + 1024) nbb = load_b_frag(ktb + kp0 + 32, K_LEN, lane);
    v16bf a = load_a_frag_lds(&Sbuf[0][0], SROW, lane, kp0);
    acc = wmma_bf16(a, bb, acc);
    bb = nbb;
  }
  {
    const int n = lane & 15;
#pragma unroll
    for (int v = 0; v < 8; ++v) Opart[half][ntile][(lane >> 4) * 8 + v][n] = acc[v];
  }
  __syncthreads();
  if (half == 0) {
    const int n = lane & 15;
    unsigned short* orow =
        obuf + ((long)b * Q_LEN + q0) * (N_HEAD * HIDDEN) + h * HIDDEN + ntile * 16;
#pragma unroll
    for (int v = 0; v < 8; ++v) {
      int m = (lane >> 4) * 8 + v;
      float val = (Opart[0][ntile][m][n] + Opart[1][ntile][m][n]) * rowinv[m];
      orow[(long)m * (N_HEAD * HIDDEN) + n] = (unsigned short)bf16_1(val);
    }
  }
}

// -------- Kernel 4: out = O @ proj_w + b.  O: [B*Q][1024] bf16, pwt: [1024 n][1024 k] bf16
__global__ __launch_bounds__(256) void proj_kernel(
    const unsigned short* __restrict__ obuf, const unsigned short* __restrict__ pwt,
    const float* __restrict__ bias, float* __restrict__ out) {
  __shared__ unsigned short lds_w[WLDS_ELEMS];  // 130 KB weight panel
  const int m0 = blockIdx.x * 128;
  const int n0 = blockIdx.y * 64;
  const int wave = threadIdx.x >> 5, lane = threadIdx.x & 31;

  stage_weights_to_lds(lds_w, pwt + (long)n0 * EMBED);

  const unsigned short* arow = obuf + (long)(m0 + wave * 16) * EMBED;
  v8f acc[4] = {};
  v16bf a = load_a_frag(arow, EMBED, lane);
  for (int k0 = 0; k0 < EMBED; k0 += 32) {
    v16bf an = a;
    if (k0 + 32 < EMBED) an = load_a_frag(arow + k0 + 32, EMBED, lane);  // pipeline
    v16bf b0 = load_b_frag_ldsw(lds_w, 0, k0, lane);
    v16bf b1 = load_b_frag_ldsw(lds_w, 16, k0, lane);
    v16bf b2 = load_b_frag_ldsw(lds_w, 32, k0, lane);
    v16bf b3 = load_b_frag_ldsw(lds_w, 48, k0, lane);
    acc[0] = wmma_bf16(a, b0, acc[0]);
    acc[1] = wmma_bf16(a, b1, acc[1]);
    acc[2] = wmma_bf16(a, b2, acc[2]);
    acc[3] = wmma_bf16(a, b3, acc[3]);
    a = an;
  }
  const int n = lane & 15;
#pragma unroll
  for (int nt = 0; nt < 4; ++nt)
#pragma unroll
    for (int v = 0; v < 8; ++v) {
      int m = (lane >> 4) * 8 + v;
      int col = n0 + nt * 16 + n;
      out[(long)(m0 + wave * 16 + m) * EMBED + col] = acc[nt][v] + bias[col];
    }
}

extern "C" void kernel_launch(void* const* d_in, const int* in_sizes, int n_in,
                              void* d_out, int out_size, void* d_ws, size_t ws_size,
                              hipStream_t stream) {
  const float* k       = (const float*)d_in[0];
  const float* q       = (const float*)d_in[1];
  const float* w_kx    = (const float*)d_in[2];
  const float* w_qx    = (const float*)d_in[3];
  const float* proj_w  = (const float*)d_in[4];
  const float* proj_b  = (const float*)d_in[5];
  const int*   mem_len = (const int*)d_in[6];

  float* out = (float*)d_out;                          // [8][512][1024] f32
  float* attn_out = out + (size_t)MB * Q_LEN * EMBED;  // [128][512][2048] f32

  size_t off = 0;
  auto alloc = [&](size_t bytes) -> void* {
    void* p = (char*)d_ws + off;
    off += (bytes + 255) & ~(size_t)255;
    return p;
  };
  unsigned short* wt_k = (unsigned short*)alloc((size_t)N_HEAD * EMBED * HIDDEN * 2);
  unsigned short* wt_q = (unsigned short*)alloc((size_t)N_HEAD * EMBED * HIDDEN * 2);
  unsigned short* pwt  = (unsigned short*)alloc((size_t)EMBED * EMBED * 2);
  unsigned short* kx   = (unsigned short*)alloc((size_t)N_HEAD * MB * K_LEN * HIDDEN * 2);
  unsigned short* kxT  = (unsigned short*)alloc((size_t)N_HEAD * MB * K_LEN * HIDDEN * 2);
  unsigned short* qxb  = (unsigned short*)alloc((size_t)N_HEAD * MB * Q_LEN * HIDDEN * 2);
  unsigned short* obuf = (unsigned short*)alloc((size_t)MB * Q_LEN * N_HEAD * HIDDEN * 2);

  // 1) weight transpose + bf16 convert
  transpose_cvt_kernel<<<dim3(1024), dim3(256), 0, stream>>>(w_kx, wt_k, N_HEAD, EMBED, HIDDEN);
  transpose_cvt_kernel<<<dim3(1024), dim3(256), 0, stream>>>(w_qx, wt_q, N_HEAD, EMBED, HIDDEN);
  transpose_cvt_kernel<<<dim3(1024), dim3(256), 0, stream>>>(proj_w, pwt, 1, EMBED, EMBED);

  // 2) kx (+ transposed copy) and qx (scale 1/sqrt(64) folded in)
  qk_proj_kernel<true><<<dim3(K_LEN / 128, MB, N_HEAD), dim3(256), 0, stream>>>(
      k, wt_k, kx, kxT, K_LEN, 1.0f);
  qk_proj_kernel<false><<<dim3(Q_LEN / 128, MB, N_HEAD), dim3(256), 0, stream>>>(
      q, wt_q, qxb, nullptr, Q_LEN, 0.125f);

  // 3) attention: scores -> LDS, softmax, attn write (non-temporal), attn @ V
  attn_kernel<<<dim3(Q_LEN / 16, MB, N_HEAD), dim3(256), 0, stream>>>(
      qxb, kx, kxT, mem_len, obuf, attn_out);

  // 4) output projection + bias
  proj_kernel<<<dim3(MB * Q_LEN / 128, EMBED / 64), dim3(256), 0, stream>>>(
      obuf, pwt, proj_b, out);
}
